// GATGRUEmbedder_32684701122857
// MI455X (gfx1250) — matrix operand order repaired and checked by reading, not compiled, not run on hardware
//
#include <hip/hip_runtime.h>
#include <math.h>

// ---- problem constants (match reference) ----
#define TT   512
#define BB   64
#define HH   1024
#define K2V  20
#define MM   (TT * BB)          // 32768 rows
#define IN0  (2 * HH + K2V + 1) // 2069
#define K0P  2080               // IN0 padded to multiple of 32
#define N3H  (3 * HH)           // 3072
#define SCAN_BLOCKS (HH / 16)   // 64

typedef __attribute__((ext_vector_type(16))) __bf16 v16bf;
typedef __attribute__((ext_vector_type(8)))  __bf16 v8bf;
typedef __attribute__((ext_vector_type(8)))  float  v8f;

union AFrag { v16bf v; v8bf h[2]; };

#define WMMA_BF16(a, b, c) \
    __builtin_amdgcn_wmma_f32_16x16x32_bf16(false, (a), false, (b), (short)0, (c), false, false)

// -------------------------------------------------------------------------
// init: zero h0 buffers + barrier state, exclusive prefix of ragged lengths
// -------------------------------------------------------------------------
__global__ void init_kernel(const int* __restrict__ lens, int* __restrict__ offs,
                            float* __restrict__ zf, __bf16* __restrict__ zb, int zn,
                            unsigned* __restrict__ barrier_state) {
    int i = blockIdx.x * blockDim.x + threadIdx.x;
    if (i < zn) { zf[i] = 0.0f; zb[i] = (__bf16)0.0f; }
    if (blockIdx.x == 0 && threadIdx.x == 0) {
        int acc = 0;
        for (int b = 0; b < BB; ++b) { offs[b] = acc; acc += lens[b]; }
        barrier_state[0] = 0u;   // arrival counter
        barrier_state[1] = 0u;   // generation
    }
}

// -------------------------------------------------------------------------
// build x = [gat_pad(1024) | t2v(21) | act_emb(1024) | zero-pad(11)] in bf16
// -------------------------------------------------------------------------
__global__ void build_x_kernel(const float* __restrict__ gat,
                               const int* __restrict__ act_seq,
                               const float* __restrict__ time_seq,
                               const int* __restrict__ lens,
                               const int* __restrict__ offs,
                               const float* __restrict__ emb,
                               const float* __restrict__ w0, const float* __restrict__ b0,
                               const float* __restrict__ wv, const float* __restrict__ bv,
                               __bf16* __restrict__ x) {
    int row = blockIdx.x;
    int t = row / BB, b = row % BB;
    float tv = time_seq[t * BB + b];
    int aid = act_seq[t * BB + b];
    bool valid = (t < lens[b]);
    long gbase = valid ? (long)(offs[b] + t) * HH : 0;
    __bf16* xr = x + (size_t)row * K0P;
    for (int c = threadIdx.x; c < K0P; c += blockDim.x) {
        float v;
        if (c < HH)                v = valid ? gat[gbase + c] : 0.0f;
        else if (c == HH)          v = tv * w0[0] + b0[0];
        else if (c < HH + 21)      v = sinf(tv * wv[c - HH - 1] + bv[c - HH - 1]);
        else if (c < HH + 21 + HH) v = emb[(long)aid * HH + (c - HH - 21)];
        else                       v = 0.0f;
        xr[c] = (__bf16)v;
    }
}

// -------------------------------------------------------------------------
// f32 -> bf16 weight convert with column zero-padding
// -------------------------------------------------------------------------
__global__ void cvt_bf16_kernel(const float* __restrict__ src, __bf16* __restrict__ dst,
                                int rows, int scols, int dcols) {
    long i = (long)blockIdx.x * blockDim.x + threadIdx.x;
    long total = (long)rows * dcols;
    if (i >= total) return;
    int r = (int)(i / dcols), c = (int)(i % dcols);
    dst[i] = (__bf16)((c < scols) ? src[(long)r * scols + c] : 0.0f);
}

// -------------------------------------------------------------------------
// C[M,N] (f32) = A[M,K] bf16 * Bw[N,K]^T bf16 + bias[N]
// 256 thr = 8 waves; block tile 128(M) x 128(N); wave tile 32(M) x 64(N)
// = 2x4 WMMA accumulators: 12 b128-loads per 8 WMMA per K-step.
// -------------------------------------------------------------------------
__global__ __launch_bounds__(256)
void gemm_bf16_wmma(const __bf16* __restrict__ A, int lda,
                    const __bf16* __restrict__ Bw, int ldb,
                    const float* __restrict__ bias,
                    float* __restrict__ C, int ldc, int Kdim) {
    int lane = threadIdx.x & 31;
    int w = threadIdx.x >> 5;
    int wm = w & 3, wn = w >> 2;            // 4 M-groups x 2 N-groups
    int Moff = blockIdx.y * 128 + wm * 32;
    int Noff = blockIdx.x * 128 + wn * 64;

    v8f acc[2][4] = {};

    int ar    = lane & 15;
    int aksel = (lane >> 4) << 4;           // A frag: +0 / +16 within K-chunk
    int bksel = (lane >> 4) << 4;           // B frag: +0 / +16 within K-chunk

    const __bf16* a0 = A + (size_t)(Moff + ar) * lda;
    const __bf16* a1 = A + (size_t)(Moff + 16 + ar) * lda;
    const __bf16* bp0 = Bw + (size_t)(Noff +  0 + ar) * ldb;
    const __bf16* bp1 = Bw + (size_t)(Noff + 16 + ar) * ldb;
    const __bf16* bp2 = Bw + (size_t)(Noff + 32 + ar) * ldb;
    const __bf16* bp3 = Bw + (size_t)(Noff + 48 + ar) * ldb;

    for (int kk = 0; kk < Kdim; kk += 32) {
        AFrag fa0, fa1;
        fa0.h[0] = *(const v8bf*)(a0 + kk + aksel);
        fa0.h[1] = *(const v8bf*)(a0 + kk + aksel + 8);
        fa1.h[0] = *(const v8bf*)(a1 + kk + aksel);
        fa1.h[1] = *(const v8bf*)(a1 + kk + aksel + 8);
        v16bf fb0 = *(const v16bf*)(bp0 + kk + bksel);
        v16bf fb1 = *(const v16bf*)(bp1 + kk + bksel);
        v16bf fb2 = *(const v16bf*)(bp2 + kk + bksel);
        v16bf fb3 = *(const v16bf*)(bp3 + kk + bksel);
        acc[0][0] = WMMA_BF16(fa0.v, fb0, acc[0][0]);
        acc[1][0] = WMMA_BF16(fa1.v, fb0, acc[1][0]);
        acc[0][1] = WMMA_BF16(fa0.v, fb1, acc[0][1]);
        acc[1][1] = WMMA_BF16(fa1.v, fb1, acc[1][1]);
        acc[0][2] = WMMA_BF16(fa0.v, fb2, acc[0][2]);
        acc[1][2] = WMMA_BF16(fa1.v, fb2, acc[1][2]);
        acc[0][3] = WMMA_BF16(fa0.v, fb3, acc[0][3]);
        acc[1][3] = WMMA_BF16(fa1.v, fb3, acc[1][3]);
    }

    int cl = lane & 15;
    int rbase = (lane >> 4) << 3;           // C frag: lane>=16 holds rows M+8..M+15
#pragma unroll
    for (int j = 0; j < 4; ++j) {
        int col = Noff + j * 16 + cl;
        float bvj = bias[col];
#pragma unroll
        for (int e = 0; e < 8; ++e) {
            int r0 = Moff + rbase + e;
            C[(size_t)r0 * ldc + col]        = acc[0][j][e] + bvj;
            C[(size_t)(r0 + 16) * ldc + col] = acc[1][j][e] + bvj;
        }
    }
}

// -------------------------------------------------------------------------
// device-wide sense-style barrier for the persistent scan kernel
// -------------------------------------------------------------------------
__device__ __forceinline__ void grid_sync(unsigned* cnt, unsigned* gen, unsigned nb) {
    __threadfence();
    __syncthreads();
    if (threadIdx.x == 0) {
        unsigned g = __hip_atomic_load(gen, __ATOMIC_RELAXED, __HIP_MEMORY_SCOPE_AGENT);
        unsigned prev = __hip_atomic_fetch_add(cnt, 1u, __ATOMIC_ACQ_REL, __HIP_MEMORY_SCOPE_AGENT);
        if (prev == nb - 1u) {
            __hip_atomic_store(cnt, 0u, __ATOMIC_RELAXED, __HIP_MEMORY_SCOPE_AGENT);
            __hip_atomic_fetch_add(gen, 1u, __ATOMIC_RELEASE, __HIP_MEMORY_SCOPE_AGENT);
        } else {
            while (__hip_atomic_load(gen, __ATOMIC_ACQUIRE, __HIP_MEMORY_SCOPE_AGENT) == g) {
                __builtin_amdgcn_s_sleep(2);
            }
        }
    }
    __syncthreads();
    __threadfence();
}

// -------------------------------------------------------------------------
// Persistent GRU scan: ONE launch runs all 512 timesteps of a layer.
// 64 blocks; block owns hidden cols [16*bx, 16*bx+16). Its 3 gate slices of
// Whh (3 x 16 x 1024 bf16 = 96 KB) are staged into LDS once (CDNA5: 320 KB
// per WGP) and re-read via ds_load for every step, removing ~6 GB of L2
// traffic across the scan. Steps are separated by a software grid barrier,
// eliminating 512 kernel-launch latencies from the critical path.
// -------------------------------------------------------------------------
__global__ __launch_bounds__(128)
void gru_scan_wmma(const float* __restrict__ gi_all,  // [T*B, 3H], includes bih
                   const __bf16* __restrict__ Whh,    // [3H, H] bf16 row-major
                   const float* __restrict__ bhh,     // [3H]
                   const float* __restrict__ zf,      // [B, H] zeros (h0)
                   const __bf16* __restrict__ zb,     // [B, H] zeros (h0)
                   float* __restrict__ ysf,           // [T*B, H] f32 out
                   __bf16* __restrict__ ysb,          // [T*B, H] bf16 out
                   unsigned* __restrict__ barrier_state) {
    __shared__ __align__(64) __bf16 smem[3 * 16 * HH];   // 96 KB

    int lane = threadIdx.x & 31;
    int wv = threadIdx.x >> 5;          // 0..3 -> batch rows 16*wv
    int Noff = blockIdx.x * 16;         // hidden-column slice
    int Moff = wv * 16;

    // stage this block's Whh slices: 3 contiguous 32 KB chunks
#pragma unroll
    for (int g = 0; g < 3; ++g) {
        const uint4* src = (const uint4*)(Whh + ((size_t)g * HH + Noff) * HH);
        uint4* dst = (uint4*)(smem + g * 16 * HH);
        for (int i = threadIdx.x; i < 16 * HH / 8; i += blockDim.x) dst[i] = src[i];
    }
    __syncthreads();

    int ar    = lane & 15;
    int aksel = (lane >> 4) << 4;
    int bksel = (lane >> 4) << 4;
    const __bf16* brl = smem + 0 * 16 * HH + (size_t)ar * HH;
    const __bf16* bzl = smem + 1 * 16 * HH + (size_t)ar * HH;
    const __bf16* bnl = smem + 2 * 16 * HH + (size_t)ar * HH;

    int cl = lane & 15;
    int rbase = (lane >> 4) << 3;
    int col = Noff + cl;
    float br_ = bhh[col], bz_ = bhh[HH + col], bn_ = bhh[2 * HH + col];

    unsigned* cnt = barrier_state;
    unsigned* gen = barrier_state + 1;

    for (int t = 0; t < TT; ++t) {
        const float*  hprev  = t ? ysf + (size_t)(t - 1) * BB * HH : zf;
        const __bf16* hprevb = t ? ysb + (size_t)(t - 1) * BB * HH : zb;
        const float*  gi     = gi_all + (size_t)t * BB * N3H;

        v8f cr = {}, cz = {}, cn = {};
        const __bf16* ap = hprevb + (size_t)(Moff + ar) * HH;
        for (int kk = 0; kk < HH; kk += 32) {
            AFrag fa;
            fa.h[0] = *(const v8bf*)(ap + kk + aksel);
            fa.h[1] = *(const v8bf*)(ap + kk + aksel + 8);
            v16bf fr = *(const v16bf*)(brl + kk + bksel);
            v16bf fz = *(const v16bf*)(bzl + kk + bksel);
            v16bf fn = *(const v16bf*)(bnl + kk + bksel);
            cr = WMMA_BF16(fa.v, fr, cr);
            cz = WMMA_BF16(fa.v, fz, cz);
            cn = WMMA_BF16(fa.v, fn, cn);
        }

#pragma unroll
        for (int e = 0; e < 8; ++e) {
            int b = Moff + rbase + e;
            const float* gib = gi + (size_t)b * N3H;
            float r = 1.0f / (1.0f + expf(-(gib[col] + cr[e] + br_)));
            float z = 1.0f / (1.0f + expf(-(gib[HH + col] + cz[e] + bz_)));
            float n = tanhf(gib[2 * HH + col] + r * (cn[e] + bn_));
            float hp = hprev[(size_t)b * HH + col];
            float h = (1.0f - z) * n + z * hp;
            ysf[(size_t)t * BB * HH + (size_t)b * HH + col] = h;
            ysb[(size_t)t * BB * HH + (size_t)b * HH + col] = (__bf16)h;
        }

        grid_sync(cnt, gen, (unsigned)gridDim.x);
    }
}

// -------------------------------------------------------------------------
extern "C" void kernel_launch(void* const* d_in, const int* in_sizes, int n_in,
                              void* d_out, int out_size, void* d_ws, size_t ws_size,
                              hipStream_t stream) {
    const float* gat   = (const float*)d_in[0];
    const int*   act   = (const int*)  d_in[1];
    const float* tim   = (const float*)d_in[2];
    const int*   lens  = (const int*)  d_in[3];
    const float* emb   = (const float*)d_in[4];
    const float* w0    = (const float*)d_in[5];
    const float* b0    = (const float*)d_in[6];
    const float* wv    = (const float*)d_in[7];
    const float* bv    = (const float*)d_in[8];
    const float* Wih0  = (const float*)d_in[9];
    const float* Whh0  = (const float*)d_in[10];
    const float* bih0  = (const float*)d_in[11];
    const float* bhh0  = (const float*)d_in[12];
    const float* Wih1  = (const float*)d_in[13];
    const float* Whh1  = (const float*)d_in[14];
    const float* bih1  = (const float*)d_in[15];
    const float* bhh1  = (const float*)d_in[16];
    (void)in_sizes; (void)n_in; (void)out_size; (void)ws_size;

    // ---- carve workspace (256B-aligned segments) ----
    char* p = (char*)d_ws;
    auto take = [&](size_t bytes) -> void* {
        void* r = (void*)p;
        p += (bytes + 255) & ~(size_t)255;
        return r;
    };
    __bf16*   xbf   = (__bf16*)  take((size_t)MM * K0P * 2);
    __bf16*   Wih0b = (__bf16*)  take((size_t)N3H * K0P * 2);
    __bf16*   Whh0b = (__bf16*)  take((size_t)N3H * HH * 2);
    __bf16*   Wih1b = (__bf16*)  take((size_t)N3H * HH * 2);
    __bf16*   Whh1b = (__bf16*)  take((size_t)N3H * HH * 2);
    float*    gi    = (float*)   take((size_t)MM * N3H * 4);   // reused by both layers
    float*    ys0   = (float*)   take((size_t)MM * HH * 4);
    __bf16*   ys0b  = (__bf16*)  take((size_t)MM * HH * 2);
    __bf16*   ys1b  = (__bf16*)  take((size_t)MM * HH * 2);
    float*    zf    = (float*)   take((size_t)BB * HH * 4);
    __bf16*   zb    = (__bf16*)  take((size_t)BB * HH * 2);
    int*      offs  = (int*)     take(BB * 4);
    unsigned* bar   = (unsigned*)take(2 * sizeof(unsigned));

    float* out = (float*)d_out;

    // ---- pipeline ----
    init_kernel<<<(BB * HH + 255) / 256, 256, 0, stream>>>(lens, offs, zf, zb, BB * HH, bar);

    build_x_kernel<<<MM, 256, 0, stream>>>(gat, act, tim, lens, offs, emb, w0, b0, wv, bv, xbf);

    {
        long n;
        n = (long)N3H * K0P;
        cvt_bf16_kernel<<<(unsigned)((n + 255) / 256), 256, 0, stream>>>(Wih0, Wih0b, N3H, IN0, K0P);
        n = (long)N3H * HH;
        cvt_bf16_kernel<<<(unsigned)((n + 255) / 256), 256, 0, stream>>>(Whh0, Whh0b, N3H, HH, HH);
        cvt_bf16_kernel<<<(unsigned)((n + 255) / 256), 256, 0, stream>>>(Wih1, Wih1b, N3H, HH, HH);
        cvt_bf16_kernel<<<(unsigned)((n + 255) / 256), 256, 0, stream>>>(Whh1, Whh1b, N3H, HH, HH);
    }

    dim3 gemm_grid(N3H / 128, MM / 128);

    // layer 0: input-gate GEMM, then persistent recurrent scan
    gemm_bf16_wmma<<<gemm_grid, 256, 0, stream>>>(xbf, K0P, Wih0b, K0P, bih0, gi, N3H, K0P);
    gru_scan_wmma<<<SCAN_BLOCKS, 128, 0, stream>>>(gi, Whh0b, bhh0, zf, zb, ys0, ys0b, bar);

    // layer 1: input-gate GEMM over layer-0 outputs, then scan -> d_out
    gemm_bf16_wmma<<<gemm_grid, 256, 0, stream>>>(ys0b, HH, Wih1b, HH, bih1, gi, N3H, HH);
    gru_scan_wmma<<<SCAN_BLOCKS, 128, 0, stream>>>(gi, Whh1b, bhh1, zf, zb, out, ys1b, bar);
}
